// GraphNeuralNetworkEncoder_57939108823845
// MI455X (gfx1250) — matrix operand order repaired and checked by reading, not compiled, not run on hardware
//
#include <hip/hip_runtime.h>

// ---------------------------------------------------------------------------
// Types for CDNA5 WMMA (wave32)
// ---------------------------------------------------------------------------
typedef __attribute__((ext_vector_type(16))) __bf16 v16bf;
typedef __attribute__((ext_vector_type(8)))  float  v8f;

#define DEVINL __device__ __forceinline__

DEVINL unsigned short bf16bits(float f) {
  unsigned u = __float_as_uint(f);
  u += 0x7FFFu + ((u >> 16) & 1u);            // round-to-nearest-even
  return (unsigned short)(u >> 16);
}

// monotonic float -> uint mapping so segment-max can use atomicMax(u32)
DEVINL unsigned fkey(float f) {
  unsigned u = __float_as_uint(f);
  return (u & 0x80000000u) ? ~u : (u | 0x80000000u);
}
DEVINL float funkey(unsigned u) {
  return __uint_as_float((u & 0x80000000u) ? (u ^ 0x80000000u) : ~u);
}

#define GS_LOOP(i, total)                                                     \
  for (long long i = (long long)blockIdx.x * blockDim.x + threadIdx.x;        \
       i < (total); i += (long long)gridDim.x * blockDim.x)

// ---------------------------------------------------------------------------
// Utility kernels
// ---------------------------------------------------------------------------
__global__ void k_zero(float* __restrict__ p, long long n) {
  GS_LOOP(i, n) p[i] = 0.0f;
}

__global__ void k_deg(const int* __restrict__ dst, float* __restrict__ deg, int E) {
  GS_LOOP(i, (long long)E) atomicAdd(deg + dst[i], 1.0f);
}

__global__ void k_dinv(const float* __restrict__ deg, float* __restrict__ dinv,
                       float* __restrict__ invdegS, int N) {
  GS_LOOP(i, (long long)N) {
    float d = deg[i];
    dinv[i]    = rsqrtf(d + 1.0f);            // GCN: +1 self loop
    invdegS[i] = 1.0f / fmaxf(d, 1.0f);       // SAGE mean
  }
}

__global__ void k_norm(const int* __restrict__ src, const int* __restrict__ dst,
                       const float* __restrict__ dinv, float* __restrict__ nrm, int E) {
  GS_LOOP(i, (long long)E) nrm[i] = dinv[src[i]] * dinv[dst[i]];
}

// out[dst] += feat[src] * (norm ? norm[e] : 1);  F = 4<<shift floats per row
__global__ void k_scatter(const float* __restrict__ feat, float* __restrict__ out,
                          const int* __restrict__ src, const int* __restrict__ dst,
                          const float* __restrict__ nrm, long long E, int shift) {
  const int F4 = 1 << shift;
  const int F  = F4 << 2;
  GS_LOOP(i, E << shift) {
    long long e = i >> shift;
    int c4 = (int)(i & (F4 - 1));
    int s = src[e], d = dst[e];
    float w = nrm ? nrm[e] : 1.0f;
    const float4 v = *(const float4*)(feat + (size_t)s * F + (c4 << 2));
    float* o = out + (size_t)d * F + (c4 << 2);
    atomicAdd(o + 0, v.x * w);
    atomicAdd(o + 1, v.y * w);
    atomicAdd(o + 2, v.z * w);
    atomicAdd(o + 3, v.w * w);
  }
}

// h = relu(agg + t*dinv^2 + b)   (self-loop folded in; out may alias t)
__global__ void k_gcn_fin(const float* __restrict__ agg, const float* t,
                          const float* __restrict__ dinv, const float* __restrict__ b,
                          float* out, int N) {
  GS_LOOP(i, (long long)N << 7) {
    int n = (int)(i >> 7), c = (int)(i & 127);
    float di = dinv[n];
    out[i] = fmaxf(agg[i] + t[i] * di * di + b[c], 0.0f);
  }
}

// ---------------------------------------------------------------------------
// bf16 conversion / packing (hoisted out of the WMMA inner loop)
// ---------------------------------------------------------------------------
// dst[i] = bf16( src[i] * (rs ? rs[i>>shift] : 1) )   -- row-scale folds SAGE mean
__global__ void k_cvt(const float* __restrict__ s, unsigned short* __restrict__ d,
                      long long n, const float* __restrict__ rs, int shift) {
  GS_LOOP(i, n) {
    float f = s[i];
    if (rs) f *= rs[i >> shift];
    d[i] = bf16bits(f);
  }
}

// Pack B[K,Nout] (row-major f32) into per-lane WMMA fragment order:
// Bp[((kt*ntn + nt)*32 + lane)*16 + s] = bf16(B[(kt*32 + (s&7) + 8*(lane>>4) + 16*(s>>3))*Nout + nt*16 + (lane&15)])
__global__ void k_packB(const float* __restrict__ B, unsigned short* __restrict__ Bp,
                        int K, int Nout) {
  const int ntn = Nout >> 4;
  GS_LOOP(i, (long long)K * Nout) {
    long long tile = i >> 9;                  // 512 packed elements per (kt,nt) tile
    int r    = (int)(i & 511);
    int lane = r >> 4;
    int idx  = r & 15;
    int v    = idx >> 1, lohi = idx & 1;
    int half = lane >> 4;
    int kt   = (int)(tile / ntn), nt = (int)(tile % ntn);
    int kB   = (idx & 7) + 8 * half + 16 * (idx >> 3);
    Bp[i] = bf16bits(B[(size_t)(kt * 32 + kB) * Nout + nt * 16 + (lane & 15)]);
  }
}

// ---------------------------------------------------------------------------
// WMMA GEMM on pre-converted bf16 A (row-major) and pre-packed B fragments.
// Inner loop: 4x global_load_b128 + 1x v_wmma_f32_16x16x32_bf16.
// ---------------------------------------------------------------------------
struct U8 { uint4 a, b; };                    // 32 bytes == v16bf

DEVINL v16bf load_frag(const unsigned short* p0, const unsigned short* p1) {
  U8 u;
  u.a = *(const uint4*)p0;
  u.b = *(const uint4*)p1;
  return __builtin_bit_cast(v16bf, u);
}

__global__ void gemm_bf16(const unsigned short* __restrict__ A1,
                          const unsigned short* __restrict__ B1p, int K1,
                          const unsigned short* __restrict__ A2,
                          const unsigned short* __restrict__ B2p, int K2,
                          const float* __restrict__ bias, float* __restrict__ C,
                          int M, int Nout, int doRelu) {
  int wave = blockIdx.x * (blockDim.x >> 5) + (threadIdx.x >> 5);
  int lane = threadIdx.x & 31;
  int ntn = Nout >> 4;
  int total = (M >> 4) * ntn;
  if (wave >= total) return;                  // wave-uniform: EXEC stays all-1 for WMMA
  int tn = wave % ntn;
  int m0 = (wave / ntn) << 4;
  int n0 = tn << 4;
  int ml = m0 + (lane & 15);                  // A row handled by this lane
  int nl = n0 + (lane & 15);                  // C column handled by this lane
  int half = lane >> 4;

  v8f c = {0.f, 0.f, 0.f, 0.f, 0.f, 0.f, 0.f, 0.f};

  for (int k0 = 0; k0 < K1; k0 += 32) {
    // A 16x32 bf16 fragment: lane holds k0+8h..+7 and k0+8h+16..+23 (two 16B runs)
    const unsigned short* ar = A1 + (size_t)ml * K1 + k0 + 8 * half;
    v16bf a = load_frag(ar, ar + 16);
    // B fragment: 16 contiguous bf16 from the packed buffer
    const unsigned short* br = B1p + (((size_t)(k0 >> 5) * ntn + tn) * 32 + lane) * 16;
    v16bf b = load_frag(br, br + 8);
    c = __builtin_amdgcn_wmma_f32_16x16x32_bf16(false, a, false, b, (short)0, c, false, false);
  }
  if (A2) {
    for (int k0 = 0; k0 < K2; k0 += 32) {
      const unsigned short* ar = A2 + (size_t)ml * K2 + k0 + 8 * half;
      v16bf a = load_frag(ar, ar + 16);
      const unsigned short* br = B2p + (((size_t)(k0 >> 5) * ntn + tn) * 32 + lane) * 16;
      v16bf b = load_frag(br, br + 8);
      c = __builtin_amdgcn_wmma_f32_16x16x32_bf16(false, a, false, b, (short)0, c, false, false);
    }
  }

  float bv = bias ? bias[nl] : 0.0f;
#pragma unroll
  for (int r = 0; r < 8; ++r) {               // C layout: VGPR r -> row r (lo half) / r+8 (hi half)
    float v = c[r] + bv;
    if (doRelu) v = fmaxf(v, 0.0f);
    C[(size_t)(m0 + r + 8 * half) * Nout + nl] = v;
  }
}

// ---------------------------------------------------------------------------
// GAT kernels
// ---------------------------------------------------------------------------
__global__ void k_attdot(const float* __restrict__ hg, const float* __restrict__ asrc,
                         const float* __restrict__ adst, float* __restrict__ als,
                         float* __restrict__ ald, int N) {
  GS_LOOP(i, (long long)N << 3) {
    int n = (int)(i >> 3), h = (int)(i & 7);
    const float4* hp = (const float4*)(hg + (size_t)n * 512 + h * 64);
    const float4* ws = (const float4*)(asrc + h * 64);
    const float4* wd = (const float4*)(adst + h * 64);
    float s = 0.f, d = 0.f;
#pragma unroll 4
    for (int j = 0; j < 16; ++j) {
      float4 v = hp[j], a = ws[j], b = wd[j];
      s += v.x * a.x + v.y * a.y + v.z * a.z + v.w * a.w;
      d += v.x * b.x + v.y * b.y + v.z * b.z + v.w * b.w;
    }
    als[i] = s;
    ald[i] = d;
  }
}

__global__ void k_gat_pass1(const int* __restrict__ src, const int* __restrict__ dst,
                            const float* __restrict__ als, const float* __restrict__ ald,
                            float* __restrict__ ews, unsigned* __restrict__ mkey,
                            int E, int N) {
  GS_LOOP(i, (long long)(E + N) << 3) {
    long long e = i >> 3;
    int h = (int)(i & 7);
    int s, d;
    if (e < E) { s = src[e]; d = dst[e]; } else { s = d = (int)(e - E); }
    float l = als[(size_t)s * 8 + h] + ald[(size_t)d * 8 + h];
    l = (l > 0.f) ? l : 0.2f * l;             // leaky relu
    ews[i] = l;
    atomicMax(mkey + (size_t)d * 8 + h, fkey(l));
  }
}

__global__ void k_gat_pass2(const int* __restrict__ dst, float* __restrict__ ews,
                            const unsigned* __restrict__ mkey, float* __restrict__ denom,
                            int E, int N) {
  GS_LOOP(i, (long long)(E + N) << 3) {
    long long e = i >> 3;
    int h = (int)(i & 7);
    int d = (e < E) ? dst[e] : (int)(e - E);
    float m = funkey(mkey[(size_t)d * 8 + h]);
    float v = __expf(ews[i] - m);
    ews[i] = v;
    atomicAdd(denom + (size_t)d * 8 + h, v);
  }
}

// gat[dst,c] += hg[src,h*64+c] * alpha / HEADS  (head-mean folded into scatter)
__global__ void k_gat_pass3(const int* __restrict__ src, const int* __restrict__ dst,
                            const float* __restrict__ ews, const float* __restrict__ denom,
                            const float* __restrict__ hg, float* __restrict__ gat,
                            int E, int N) {
  GS_LOOP(i, (long long)(E + N) << 7) {
    long long e = i >> 7;
    int r = (int)(i & 127);
    int h = r >> 4, c4 = (r & 15) << 2;
    int s, d;
    if (e < E) { s = src[e]; d = dst[e]; } else { s = d = (int)(e - E); }
    float w = ews[e * 8 + h] / denom[(size_t)d * 8 + h] * 0.125f;
    const float4 v = *(const float4*)(hg + (size_t)s * 512 + h * 64 + c4);
    float* o = gat + (size_t)d * 64 + c4;
    atomicAdd(o + 0, v.x * w);
    atomicAdd(o + 1, v.y * w);
    atomicAdd(o + 2, v.z * w);
    atomicAdd(o + 3, v.w * w);
  }
}

__global__ void k_addbias64(float* __restrict__ p, const float* __restrict__ b, int N) {
  GS_LOOP(i, (long long)N << 6) p[i] += b[i & 63];
}

// ---------------------------------------------------------------------------
// Pooling
// ---------------------------------------------------------------------------
__global__ void k_cnt(const int* __restrict__ batch, float* __restrict__ cnt, int N) {
  GS_LOOP(i, (long long)N) atomicAdd(cnt + batch[i], 1.0f);
}
__global__ void k_poolsum(const int* __restrict__ batch, const float* __restrict__ emb,
                          float* __restrict__ out, int N) {
  GS_LOOP(i, (long long)N << 6) {
    int n = (int)(i >> 6), c = (int)(i & 63);
    atomicAdd(out + (size_t)batch[n] * 64 + c, emb[i]);
  }
}
__global__ void k_pooldiv(float* __restrict__ out, const float* __restrict__ cnt, int G) {
  GS_LOOP(i, (long long)G << 6) out[i] /= fmaxf(cnt[i >> 6], 1.0f);
}

// ---------------------------------------------------------------------------
// Host orchestration
// ---------------------------------------------------------------------------
extern "C" void kernel_launch(void* const* d_in, const int* in_sizes, int n_in,
                              void* d_out, int out_size, void* d_ws, size_t ws_size,
                              hipStream_t stream) {
  (void)in_sizes; (void)n_in; (void)out_size; (void)ws_size;
  const int N = 20000, E = 320000, G = 64;

  const float* x    = (const float*)d_in[0];
  const int*   ei   = (const int*)d_in[1];
  const int*   srcA = ei;
  const int*   dstA = ei + E;
  const int*   bat  = (const int*)d_in[2];
  const float* w0   = (const float*)d_in[3];
  const float* b0   = (const float*)d_in[4];
  const float* w1   = (const float*)d_in[5];
  const float* b1   = (const float*)d_in[6];
  const float* gw   = (const float*)d_in[7];
  const float* gas  = (const float*)d_in[8];
  const float* gad  = (const float*)d_in[9];
  const float* gbv  = (const float*)d_in[10];
  const float* s1wl = (const float*)d_in[11];
  const float* s1wr = (const float*)d_in[12];
  const float* s1b  = (const float*)d_in[13];
  const float* s2wl = (const float*)d_in[14];
  const float* s2wr = (const float*)d_in[15];
  const float* s2b  = (const float*)d_in[16];
  const float* pw   = (const float*)d_in[17];
  const float* pb   = (const float*)d_in[18];
  float* out = (float*)d_out;

  float* ws = (float*)d_ws;
  size_t off = 0;
  auto alloc = [&](size_t n) -> float* {
    float* p = ws + off;
    off += (n + 1023) & ~(size_t)1023;
    return p;
  };
  auto allocUS = [&](size_t n) -> unsigned short* {   // n ushorts
    return (unsigned short*)alloc((n + 1) / 2);
  };
  float* bufA    = alloc((size_t)N * 128);     // h (node features)
  float* bufT    = alloc((size_t)N * 512);     // GEMM temp / hg
  float* bufB    = alloc((size_t)N * 128);     // agg scratch / emb
  float* als     = alloc((size_t)N * 8);
  float* ald     = alloc((size_t)N * 8);
  float* mkey    = alloc((size_t)N * 8);       // uint keys
  float* denom   = alloc((size_t)N * 8);
  float* ews     = alloc((size_t)(E + N) * 8); // edge norms, then GAT logits/exp
  float* gat     = alloc((size_t)N * 64);
  float* s1      = alloc((size_t)N * 64);
  float* agg64   = alloc((size_t)N * 64);
  float* s2      = alloc((size_t)N * 64);
  float* deg     = alloc(N);
  float* dinv    = alloc(N);
  float* invdegS = alloc(N);
  float* cnt     = alloc(G);
  float* emb     = bufB;
  // bf16 staging
  unsigned short* bfP   = allocUS((size_t)N * 128);
  unsigned short* bfQ   = allocUS((size_t)N * 128);
  unsigned short* bfH   = allocUS((size_t)N * 128);   // persistent bf16 copy of h2
  unsigned short* w0p   = allocUS(128 * 128);
  unsigned short* w1p   = allocUS(128 * 128);
  unsigned short* gwp   = allocUS(128 * 512);
  unsigned short* s1wlp = allocUS(128 * 64);
  unsigned short* s1wrp = allocUS(128 * 64);
  unsigned short* s2wlp = allocUS(64 * 64);
  unsigned short* s2wrp = allocUS(64 * 64);
  unsigned short* pw1p  = allocUS(64 * 64);
  unsigned short* pw2p  = allocUS(64 * 64);

  auto nb = [](long long w) -> int {
    long long b = (w + 255) / 256;
    return (int)(b > 16384 ? 16384 : b);
  };
  auto gnb = [](int M, int Nout) -> int { return ((M / 16) * (Nout / 16) + 7) / 8; };

  // weight packing (tiny)
  k_packB<<<nb(128 * 128), 256, 0, stream>>>(w0, w0p, 128, 128);
  k_packB<<<nb(128 * 128), 256, 0, stream>>>(w1, w1p, 128, 128);
  k_packB<<<nb(128 * 512), 256, 0, stream>>>(gw, gwp, 128, 512);
  k_packB<<<nb(128 * 64), 256, 0, stream>>>(s1wl, s1wlp, 128, 64);
  k_packB<<<nb(128 * 64), 256, 0, stream>>>(s1wr, s1wrp, 128, 64);
  k_packB<<<nb(64 * 64), 256, 0, stream>>>(s2wl, s2wlp, 64, 64);
  k_packB<<<nb(64 * 64), 256, 0, stream>>>(s2wr, s2wrp, 64, 64);
  k_packB<<<nb(64 * 64), 256, 0, stream>>>(pw, pw1p, 64, 64);            // rows 0..63
  k_packB<<<nb(64 * 64), 256, 0, stream>>>(pw + 64 * 64, pw2p, 64, 64);  // rows 64..127

  // degrees / norms
  k_zero<<<nb(N), 256, 0, stream>>>(deg, N);
  k_deg<<<nb(E), 256, 0, stream>>>(dstA, deg, E);
  k_dinv<<<nb(N), 256, 0, stream>>>(deg, dinv, invdegS, N);
  k_norm<<<nb(E), 256, 0, stream>>>(srcA, dstA, dinv, ews, E);

  // GCN layer 0: t = x@W0 ; agg ; h = relu(agg + t*dinv^2 + b0)
  k_cvt<<<nb((long long)N * 128), 256, 0, stream>>>(x, bfP, (long long)N * 128, nullptr, 0);
  gemm_bf16<<<gnb(N, 128), 256, 0, stream>>>(bfP, w0p, 128, nullptr, nullptr, 0,
                                             nullptr, bufA, N, 128, 0);
  k_zero<<<nb((long long)N * 128), 256, 0, stream>>>(bufB, (long long)N * 128);
  k_scatter<<<nb((long long)E * 32), 256, 0, stream>>>(bufA, bufB, srcA, dstA, ews, E, 5);
  k_gcn_fin<<<nb((long long)N * 128), 256, 0, stream>>>(bufB, bufA, dinv, b0, bufA, N);

  // GCN layer 1
  k_cvt<<<nb((long long)N * 128), 256, 0, stream>>>(bufA, bfP, (long long)N * 128, nullptr, 0);
  gemm_bf16<<<gnb(N, 128), 256, 0, stream>>>(bfP, w1p, 128, nullptr, nullptr, 0,
                                             nullptr, bufT, N, 128, 0);
  k_zero<<<nb((long long)N * 128), 256, 0, stream>>>(bufB, (long long)N * 128);
  k_scatter<<<nb((long long)E * 32), 256, 0, stream>>>(bufT, bufB, srcA, dstA, ews, E, 5);
  k_gcn_fin<<<nb((long long)N * 128), 256, 0, stream>>>(bufB, bufT, dinv, b1, bufA, N);

  // GAT
  k_cvt<<<nb((long long)N * 128), 256, 0, stream>>>(bufA, bfH, (long long)N * 128, nullptr, 0);
  gemm_bf16<<<gnb(N, 512), 256, 0, stream>>>(bfH, gwp, 128, nullptr, nullptr, 0,
                                             nullptr, bufT, N, 512, 0);
  k_attdot<<<nb((long long)N * 8), 256, 0, stream>>>(bufT, gas, gad, als, ald, N);
  k_zero<<<nb((long long)N * 8), 256, 0, stream>>>(mkey, (long long)N * 8);   // 0 == -inf key
  k_zero<<<nb((long long)N * 8), 256, 0, stream>>>(denom, (long long)N * 8);
  k_zero<<<nb((long long)N * 64), 256, 0, stream>>>(gat, (long long)N * 64);
  k_gat_pass1<<<nb((long long)(E + N) * 8), 256, 0, stream>>>(srcA, dstA, als, ald, ews,
                                                              (unsigned*)mkey, E, N);
  k_gat_pass2<<<nb((long long)(E + N) * 8), 256, 0, stream>>>(dstA, ews, (unsigned*)mkey,
                                                              denom, E, N);
  k_gat_pass3<<<nb((long long)(E + N) * 128), 256, 0, stream>>>(srcA, dstA, ews, denom,
                                                                bufT, gat, E, N);
  k_addbias64<<<nb((long long)N * 64), 256, 0, stream>>>(gat, gbv, N);

  // SAGE 1: s1 = relu(mean_agg(h)@Wl + h@Wr + b)   (mean folded into cvt row-scale)
  k_zero<<<nb((long long)N * 128), 256, 0, stream>>>(bufB, (long long)N * 128);
  k_scatter<<<nb((long long)E * 32), 256, 0, stream>>>(bufA, bufB, srcA, dstA, nullptr, E, 5);
  k_cvt<<<nb((long long)N * 128), 256, 0, stream>>>(bufB, bfP, (long long)N * 128, invdegS, 7);
  gemm_bf16<<<gnb(N, 64), 256, 0, stream>>>(bfP, s1wlp, 128, bfH, s1wrp, 128,
                                            s1b, s1, N, 64, 1);
  // SAGE 2
  k_zero<<<nb((long long)N * 64), 256, 0, stream>>>(agg64, (long long)N * 64);
  k_scatter<<<nb((long long)E * 16), 256, 0, stream>>>(s1, agg64, srcA, dstA, nullptr, E, 4);
  k_cvt<<<nb((long long)N * 64), 256, 0, stream>>>(agg64, bfP, (long long)N * 64, invdegS, 6);
  k_cvt<<<nb((long long)N * 64), 256, 0, stream>>>(s1, bfQ, (long long)N * 64, nullptr, 0);
  gemm_bf16<<<gnb(N, 64), 256, 0, stream>>>(bfP, s2wlp, 64, bfQ, s2wrp, 64,
                                            s2b, s2, N, 64, 0);

  // projection of concat([gat, s2]) without materializing the concat
  k_cvt<<<nb((long long)N * 64), 256, 0, stream>>>(gat, bfP, (long long)N * 64, nullptr, 0);
  k_cvt<<<nb((long long)N * 64), 256, 0, stream>>>(s2, bfQ, (long long)N * 64, nullptr, 0);
  gemm_bf16<<<gnb(N, 64), 256, 0, stream>>>(bfP, pw1p, 64, bfQ, pw2p, 64,
                                            pb, emb, N, 64, 0);

  // global mean pool
  k_zero<<<nb((long long)G * 64), 256, 0, stream>>>(out, (long long)G * 64);
  k_zero<<<nb(G), 256, 0, stream>>>(cnt, G);
  k_cnt<<<nb(N), 256, 0, stream>>>(bat, cnt, N);
  k_poolsum<<<nb((long long)N * 64), 256, 0, stream>>>(bat, emb, out, N);
  k_pooldiv<<<nb((long long)G * 64), 256, 0, stream>>>(out, cnt, G);
}
